// DecoderLayer_49572512530892
// MI455X (gfx1250) — matrix operand order repaired and checked by reading, not compiled, not run on hardware
//
#include <hip/hip_runtime.h>

// ---------------------------------------------------------------------------
// CDNA5 (gfx1250) decoder layer: all GEMMs + attention via v_wmma_f32_16x16x32_bf16
// ---------------------------------------------------------------------------

typedef __attribute__((ext_vector_type(16))) __bf16 v16bf;
typedef __attribute__((ext_vector_type(8)))  __bf16 v8bf;
typedef __attribute__((ext_vector_type(8)))  float  v8f;

static __device__ inline v16bf cat8(v8bf lo, v8bf hi) {
  v16bf r;
#pragma unroll
  for (int i = 0; i < 8; ++i) { r[i] = lo[i]; r[8 + i] = hi[i]; }
  return r;
}

static __device__ inline v8f zero8() {
  v8f z;
#pragma unroll
  for (int i = 0; i < 8; ++i) z[i] = 0.0f;
  return z;
}

// ---------------------------------------------------------------------------
// f32 -> bf16 convert (row-major, used for activations entering WMMA A-side)
// ---------------------------------------------------------------------------
__global__ void cvt_bf16_kernel(const float* __restrict__ in,
                                __bf16* __restrict__ out, int n) {
  int i = blockIdx.x * blockDim.x + threadIdx.x;
  int stride = gridDim.x * blockDim.x;
  for (; i < n; i += stride) out[i] = (__bf16)in[i];
}

// ---------------------------------------------------------------------------
// Weight transpose+convert: W[K][N] f32 -> Wt[N][K] bf16 (WMMA B-side wants
// per-lane contiguous K at fixed output column)
// ---------------------------------------------------------------------------
__global__ __launch_bounds__(256) void wt_bf16_kernel(
    const float* __restrict__ in, __bf16* __restrict__ out, int K, int N) {
  __shared__ float tile[32][33];
  int n0 = blockIdx.x * 32, k0 = blockIdx.y * 32;
  int tx = threadIdx.x, ty = threadIdx.y;
#pragma unroll
  for (int j = ty; j < 32; j += 8)
    tile[j][tx] = in[(size_t)(k0 + j) * N + (n0 + tx)];
  __syncthreads();
#pragma unroll
  for (int j = ty; j < 32; j += 8)
    out[(size_t)(n0 + j) * K + (k0 + tx)] = (__bf16)tile[tx][j];
}

// ---------------------------------------------------------------------------
// Tiled WMMA GEMM: C[M,N] = A[M,K](bf16) * Bt[N,K](bf16)^T + bias (+residual)
// Block tile 128x128, 8 waves of 32 lanes, wave tile 32x64 (2x4 16x16 frags),
// K step 32 through LDS (stride 40 bf16 to de-conflict banks).
// Epilogue modes: 0=f32 row-major (+residual), 1=bf16 row-major (+relu),
//                 2=bf16 [B,H,T,64] (Q/K layout), 3=bf16 [B,H,64,T] (V^T).
// ---------------------------------------------------------------------------
#define LDT 40

__global__ __launch_bounds__(256) void gemm_bf16_kernel(
    const __bf16* __restrict__ A, const __bf16* __restrict__ Bt,
    const float* __restrict__ bias, const float* __restrict__ residual,
    float* __restrict__ outF, __bf16* __restrict__ outB,
    int M, int N, int K, int mode, int relu, int Tlen) {
  __shared__ __bf16 sA[128 * LDT];
  __shared__ __bf16 sB[128 * LDT];
  int tid  = threadIdx.x;
  int lane = tid & 31;
  int wave = tid >> 5;
  int mW = wave >> 1, nW = wave & 1;
  int m0 = blockIdx.y * 128, n0 = blockIdx.x * 128;
  int lh = lane >> 4, ll = lane & 15;

  v8f acc[2][4];
#pragma unroll
  for (int i = 0; i < 2; ++i)
#pragma unroll
    for (int j = 0; j < 4; ++j) acc[i][j] = zero8();

  int kTiles = K / 32;
  for (int kt = 0; kt < kTiles; ++kt) {
#pragma unroll
    for (int r = 0; r < 2; ++r) {
      int id  = tid + r * 256;     // 0..511
      int row = id >> 2;           // 0..127
      int cc  = id & 3;            // 4 x 8 bf16 per row
      *(uint4*)(&sA[row * LDT + cc * 8]) =
          *(const uint4*)(A + (size_t)(m0 + row) * K + kt * 32 + cc * 8);
      *(uint4*)(&sB[row * LDT + cc * 8]) =
          *(const uint4*)(Bt + (size_t)(n0 + row) * K + kt * 32 + cc * 8);
    }
    if (kt + 1 < kTiles) {
      __builtin_prefetch(A + (size_t)(m0 + (tid >> 1)) * K + (kt + 1) * 32, 0, 0);
      __builtin_prefetch(Bt + (size_t)(n0 + (tid >> 1)) * K + (kt + 1) * 32, 0, 0);
    }
    __syncthreads();

    v16bf aF[2], bF[4];
#pragma unroll
    for (int mf = 0; mf < 2; ++mf) {
      int row = mW * 32 + mf * 16 + ll;
      int k0  = lh * 8;                    // lanes 0-15: K 0..7/16..23; 16-31: 8..15/24..31
      v8bf lo = *(const v8bf*)(&sA[row * LDT + k0]);
      v8bf hi = *(const v8bf*)(&sA[row * LDT + k0 + 16]);
      aF[mf] = cat8(lo, hi);
    }
#pragma unroll
    for (int nf = 0; nf < 4; ++nf) {
      int row = nW * 64 + nf * 16 + ll;
      int k0  = lh * 8;                    // lanes 0-15: K 0..7/16..23; 16-31: 8..15/24..31
      v8bf lo = *(const v8bf*)(&sB[row * LDT + k0]);
      v8bf hi = *(const v8bf*)(&sB[row * LDT + k0 + 16]);
      bF[nf] = cat8(lo, hi);
    }
#pragma unroll
    for (int mf = 0; mf < 2; ++mf)
#pragma unroll
      for (int nf = 0; nf < 4; ++nf)
        acc[mf][nf] = __builtin_amdgcn_wmma_f32_16x16x32_bf16(
            false, aF[mf], false, bF[nf], (short)0, acc[mf][nf], false, false);
    __syncthreads();
  }

  // epilogue: C layout -> lane = column ll, VGPR r = row (lanes>=16: row+8)
#pragma unroll
  for (int mf = 0; mf < 2; ++mf)
#pragma unroll
    for (int nf = 0; nf < 4; ++nf)
#pragma unroll
      for (int r = 0; r < 8; ++r) {
        int rg = m0 + mW * 32 + mf * 16 + lh * 8 + r;
        int cg = n0 + nW * 64 + nf * 16 + ll;
        float v = acc[mf][nf][r] + bias[cg];
        if (residual) v += residual[(size_t)rg * N + cg];
        if (relu) v = v > 0.0f ? v : 0.0f;
        if (mode == 0) {
          outF[(size_t)rg * N + cg] = v;
        } else if (mode == 1) {
          outB[(size_t)rg * N + cg] = (__bf16)v;
        } else {
          int bb = rg / Tlen, t = rg % Tlen;
          int h = cg >> 6, d = cg & 63;
          if (mode == 2)
            outB[((size_t)(bb * 16 + h) * Tlen + t) * 64 + d] = (__bf16)v;
          else
            outB[((size_t)(bb * 16 + h) * 64 + d) * Tlen + t] = (__bf16)v;
        }
      }
}

// ---------------------------------------------------------------------------
// Flash attention (wave32): each wave owns one 16-row q tile; online softmax
// with 16-lane shfl_xor row reductions (C layout: lane = col, VGPR = row).
// Q,K: [B,H,T,64] bf16 ; Vt: [B,H,64,S] bf16 ; ctx out: [B,T,1024] bf16.
// ---------------------------------------------------------------------------
__global__ __launch_bounds__(256) void attn_kernel(
    const __bf16* __restrict__ Q, const __bf16* __restrict__ Kb,
    const __bf16* __restrict__ Vt, __bf16* __restrict__ ctx,
    int Tq, int Skv, int causal) {
  __shared__ __bf16 stage[8][16 * LDT];
  int lane = threadIdx.x & 31;
  int wave = threadIdx.x >> 5;
  int lh = lane >> 4, ll = lane & 15;
  int bh = blockIdx.y;
  int q0 = blockIdx.x * 128 + wave * 16;
  const float scale = 0.125f;  // 1/sqrt(64)
  const __bf16* Qh = Q + (size_t)bh * Tq * 64;
  const __bf16* Kh = Kb + (size_t)bh * Skv * 64;
  const __bf16* Vh = Vt + (size_t)bh * 64 * Skv;

  v16bf aQ[2];
#pragma unroll
  for (int kc = 0; kc < 2; ++kc) {
    int row = q0 + ll;
    int k0  = kc * 32 + lh * 8;
    v8bf lo = *(const v8bf*)(Qh + (size_t)row * 64 + k0);
    v8bf hi = *(const v8bf*)(Qh + (size_t)row * 64 + k0 + 16);
    aQ[kc] = cat8(lo, hi);
  }

  v8f acc[4];
#pragma unroll
  for (int nf = 0; nf < 4; ++nf) acc[nf] = zero8();
  float mrow[8], lrow[8];
#pragma unroll
  for (int r = 0; r < 8; ++r) { mrow[r] = -1e30f; lrow[r] = 0.0f; }

  int rowg = q0 + lh * 8;
  int sLim = causal ? (q0 + 16) : Skv;

  for (int s0 = 0; s0 < sLim; s0 += 32) {
    // scores S = Q K^T for this 16x32 tile (two 16x16 sub-tiles)
    v8f st[2];
#pragma unroll
    for (int sc = 0; sc < 2; ++sc) {
      v8f c = zero8();
#pragma unroll
      for (int kc = 0; kc < 2; ++kc) {
        int srow = s0 + sc * 16 + ll;
        int k0   = kc * 32 + lh * 8;
        v8bf lo = *(const v8bf*)(Kh + (size_t)srow * 64 + k0);
        v8bf hi = *(const v8bf*)(Kh + (size_t)srow * 64 + k0 + 16);
        c = __builtin_amdgcn_wmma_f32_16x16x32_bf16(
            false, aQ[kc], false, cat8(lo, hi), (short)0, c, false, false);
      }
      st[sc] = c;
    }
    float sv[2][8];
#pragma unroll
    for (int sc = 0; sc < 2; ++sc)
#pragma unroll
      for (int r = 0; r < 8; ++r) {
        float v = st[sc][r] * scale;
        if (causal && (s0 + sc * 16 + ll > rowg + r)) v = -1e30f;
        sv[sc][r] = v;
      }
    // online softmax per row (16-lane reductions stay in each wave half)
#pragma unroll
    for (int r = 0; r < 8; ++r) {
      float mx = fmaxf(sv[0][r], sv[1][r]);
#pragma unroll
      for (int off = 8; off > 0; off >>= 1) mx = fmaxf(mx, __shfl_xor(mx, off, 32));
      float mn = fmaxf(mrow[r], mx);
      float alpha = __expf(mrow[r] - mn);
      float p0 = __expf(sv[0][r] - mn);
      float p1 = __expf(sv[1][r] - mn);
      sv[0][r] = p0; sv[1][r] = p1;
      float ls = p0 + p1;
#pragma unroll
      for (int off = 8; off > 0; off >>= 1) ls += __shfl_xor(ls, off, 32);
      lrow[r] = lrow[r] * alpha + ls;
      mrow[r] = mn;
#pragma unroll
      for (int nf = 0; nf < 4; ++nf) acc[nf][r] *= alpha;
    }
    // round-trip P through per-wave LDS: C layout -> A layout
    __bf16* sw = stage[wave];
#pragma unroll
    for (int sc = 0; sc < 2; ++sc)
#pragma unroll
      for (int r = 0; r < 8; ++r)
        sw[(lh * 8 + r) * LDT + sc * 16 + ll] = (__bf16)sv[sc][r];
    v8bf plo = *(const v8bf*)(&sw[ll * LDT + lh * 8]);
    v8bf phi = *(const v8bf*)(&sw[ll * LDT + lh * 8 + 16]);
    v16bf aP = cat8(plo, phi);
    // O += P V : B-frag columns = head dims, K = 32 s positions (V^T layout)
#pragma unroll
    for (int nf = 0; nf < 4; ++nf) {
      int drow = nf * 16 + ll;
      int k0   = lh * 8;
      v8bf blo = *(const v8bf*)(Vh + (size_t)drow * Skv + s0 + k0);
      v8bf bhi = *(const v8bf*)(Vh + (size_t)drow * Skv + s0 + k0 + 16);
      acc[nf] = __builtin_amdgcn_wmma_f32_16x16x32_bf16(
          false, aP, false, cat8(blo, bhi), (short)0, acc[nf], false, false);
    }
  }

  int b = bh >> 4, h = bh & 15;
#pragma unroll
  for (int r = 0; r < 8; ++r) {
    float inv = 1.0f / lrow[r];
    int rg = rowg + r;
#pragma unroll
    for (int nf = 0; nf < 4; ++nf) {
      int d = nf * 16 + ll;
      ctx[(size_t)(b * Tq + rg) * 1024 + h * 64 + d] = (__bf16)(acc[nf][r] * inv);
    }
  }
}

// ---------------------------------------------------------------------------
// LayerNorm over last dim (N=1024): one block per row; optional bf16 copy for
// the next WMMA consumer.
// ---------------------------------------------------------------------------
__global__ __launch_bounds__(256) void ln_kernel(
    const float* __restrict__ in, const float* __restrict__ g,
    const float* __restrict__ b, float* __restrict__ outF,
    __bf16* __restrict__ outB, int N) {
  int row = blockIdx.x;
  const float* x = in + (size_t)row * N;
  float s = 0.0f, s2 = 0.0f;
  for (int i = threadIdx.x; i < N; i += 256) {
    float v = x[i]; s += v; s2 += v * v;
  }
  __shared__ float rs[256], rq[256];
  rs[threadIdx.x] = s; rq[threadIdx.x] = s2;
  __syncthreads();
  for (int off = 128; off > 0; off >>= 1) {
    if (threadIdx.x < off) {
      rs[threadIdx.x] += rs[threadIdx.x + off];
      rq[threadIdx.x] += rq[threadIdx.x + off];
    }
    __syncthreads();
  }
  float mu = rs[0] / N;
  float var = rq[0] / N - mu * mu;
  float rstd = rsqrtf(var + 1e-5f);
  for (int i = threadIdx.x; i < N; i += 256) {
    float v = (x[i] - mu) * rstd * g[i] + b[i];
    outF[(size_t)row * N + i] = v;
    if (outB) outB[(size_t)row * N + i] = (__bf16)v;
  }
}

// ---------------------------------------------------------------------------
// Host orchestration
// ---------------------------------------------------------------------------
extern "C" void kernel_launch(void* const* d_in, const int* in_sizes, int n_in,
                              void* d_out, int out_size, void* d_ws, size_t ws_size,
                              hipStream_t stream) {
  (void)in_sizes; (void)n_in; (void)out_size; (void)ws_size;
  const int B = 4, T = 1024, S = 1024, D = 1024, F = 4096;
  const int M = B * T;  // 4096 rows

  const float* target = (const float*)d_in[0];
  const float* source = (const float*)d_in[1];
  // d_in[2] = pad_masked (causal) -- recomputed analytically in-kernel
  const float* Wf[10] = {
      (const float*)d_in[3],  (const float*)d_in[5],  (const float*)d_in[7],
      (const float*)d_in[9],  (const float*)d_in[11], (const float*)d_in[13],
      (const float*)d_in[15], (const float*)d_in[17], (const float*)d_in[19],
      (const float*)d_in[21]};
  const float* bs[10] = {
      (const float*)d_in[4],  (const float*)d_in[6],  (const float*)d_in[8],
      (const float*)d_in[10], (const float*)d_in[12], (const float*)d_in[14],
      (const float*)d_in[16], (const float*)d_in[18], (const float*)d_in[20],
      (const float*)d_in[22]};
  const float* ln1g = (const float*)d_in[23]; const float* ln1b = (const float*)d_in[24];
  const float* ln2g = (const float*)d_in[25]; const float* ln2b = (const float*)d_in[26];
  const float* ln3g = (const float*)d_in[27]; const float* ln3b = (const float*)d_in[28];

  char* p = (char*)d_ws;
  auto alloc = [&](size_t bytes) -> void* {
    void* r = (void*)p;
    p += (bytes + 255) & ~(size_t)255;
    return r;
  };
  __bf16* wt[10];
  for (int i = 0; i < 8; ++i) wt[i] = (__bf16*)alloc((size_t)D * D * 2);
  wt[8] = (__bf16*)alloc((size_t)F * D * 2);   // W1t: [4096][1024]
  wt[9] = (__bf16*)alloc((size_t)D * F * 2);   // W2t: [1024][4096]
  __bf16* tgtB = (__bf16*)alloc((size_t)M * D * 2);
  __bf16* srcB = (__bf16*)alloc((size_t)M * D * 2);
  __bf16* Qb   = (__bf16*)alloc((size_t)M * D * 2);
  __bf16* Kbuf = (__bf16*)alloc((size_t)M * D * 2);
  __bf16* Vtb  = (__bf16*)alloc((size_t)M * D * 2);
  __bf16* ctxB = (__bf16*)alloc((size_t)M * D * 2);
  float*  resF = (float*)alloc((size_t)M * D * 4);
  float*  xF   = (float*)alloc((size_t)M * D * 4);
  __bf16* xB   = (__bf16*)alloc((size_t)M * D * 2);
  float*  yF   = (float*)alloc((size_t)M * D * 4);
  __bf16* yB   = (__bf16*)alloc((size_t)M * D * 2);
  __bf16* hB   = (__bf16*)alloc((size_t)M * F * 2);

  dim3 tblk(32, 8);
  // weights -> bf16 transposed [N][K]
  for (int i = 0; i < 8; ++i)
    wt_bf16_kernel<<<dim3(D / 32, D / 32), tblk, 0, stream>>>(Wf[i], wt[i], D, D);
  wt_bf16_kernel<<<dim3(F / 32, D / 32), tblk, 0, stream>>>(Wf[8], wt[8], D, F);
  wt_bf16_kernel<<<dim3(D / 32, F / 32), tblk, 0, stream>>>(Wf[9], wt[9], F, D);
  // activations -> bf16
  cvt_bf16_kernel<<<4096, 256, 0, stream>>>(target, tgtB, M * D);
  cvt_bf16_kernel<<<4096, 256, 0, stream>>>(source, srcB, M * D);

  auto gemm = [&](const __bf16* A, const __bf16* Bt, const float* bias,
                  const float* resid, float* oF, __bf16* oB, int Mm, int Nn,
                  int Kk, int mode, int relu, int Tlen) {
    dim3 g(Nn / 128, Mm / 128);
    gemm_bf16_kernel<<<g, 256, 0, stream>>>(A, Bt, bias, resid, oF, oB, Mm, Nn,
                                            Kk, mode, relu, Tlen);
  };

  // ---- self-attention block ----
  gemm(tgtB, wt[0], bs[0], nullptr, nullptr, Qb,   M, D, D, 2, 0, T);  // Q [B,H,T,64]
  gemm(tgtB, wt[1], bs[1], nullptr, nullptr, Kbuf, M, D, D, 2, 0, T);  // K [B,H,S,64]
  gemm(tgtB, wt[2], bs[2], nullptr, nullptr, Vtb,  M, D, D, 3, 0, T);  // V^T [B,H,64,S]
  attn_kernel<<<dim3(T / 128, B * 16), 256, 0, stream>>>(Qb, Kbuf, Vtb, ctxB, T, T, 1);
  gemm(ctxB, wt[3], bs[3], target, resF, nullptr, M, D, D, 0, 0, T);   // o-proj + residual
  ln_kernel<<<M, 256, 0, stream>>>(resF, ln1g, ln1b, xF, xB, D);

  // ---- cross-attention block ----
  gemm(xB,   wt[4], bs[4], nullptr, nullptr, Qb,   M, D, D, 2, 0, T);
  gemm(srcB, wt[5], bs[5], nullptr, nullptr, Kbuf, M, D, D, 2, 0, S);
  gemm(srcB, wt[6], bs[6], nullptr, nullptr, Vtb,  M, D, D, 3, 0, S);
  attn_kernel<<<dim3(T / 128, B * 16), 256, 0, stream>>>(Qb, Kbuf, Vtb, ctxB, T, S, 0);
  gemm(ctxB, wt[7], bs[7], xF, resF, nullptr, M, D, D, 0, 0, T);
  ln_kernel<<<M, 256, 0, stream>>>(resF, ln2g, ln2b, yF, yB, D);

  // ---- FFN block ----
  gemm(yB, wt[8], bs[8], nullptr, nullptr, hB, M, F, D, 1, 1, T);      // relu -> bf16
  gemm(hB, wt[9], bs[9], yF, resF, nullptr, M, D, F, 0, 0, T);
  ln_kernel<<<M, 256, 0, stream>>>(resF, ln3g, ln3b, (float*)d_out, nullptr, D);
}
